// DeformableSqueezeAttention_62259845922820
// MI455X (gfx1250) — matrix operand order repaired and checked by reading, not compiled, not run on hardware
//
#include <hip/hip_runtime.h>
#include <hip/hip_bf16.h>
#include <cstdint>

// ---------------------------------------------------------------------------
// DeformableSqueezeAttention for MI455X (gfx1250, wave32, WMMA).
//
// Pipeline:
//   prep:   W_val/W_off/W_attn/W_out -> transposed bf16 (B operands)
//           q = query + query_pos -> bf16 ; rp2 from reference_points
//   gemm1:  v = squeeze(value) @ W_val + b_val      (299520x256x256, bf16 out)
//           squeeze row-permutation fused into the A-tile LDS stager
//   gemm2:  off = q @ W_off + b_off                 (8192x256x256, f32 out)
//   gemm3:  logits = q @ W_attn + b_attn            (8192x256x128, f32 out)
//   sample: per (q,h) wave: softmax(16) + 16 bilinear samples from bf16 v
//   gemm4:  out = sampled @ W_out + b_out + query   (8192x256x256, f32 out)
//
// All GEMMs run on v_wmma_f32_16x16x32_bf16 (16x16 tiles, K=32/instr).
// v is kept bf16 (153 MB) so level-0 (134 MB) is ~resident in the 192 MB L2
// for the gather phase.
// ---------------------------------------------------------------------------

typedef __attribute__((ext_vector_type(16))) __bf16 v16bf;
typedef __attribute__((ext_vector_type(8)))  __bf16 v8bf;
typedef __attribute__((ext_vector_type(8)))  float  v8f;

#define NQ      8192
#define E_DIM   256
#define NHEAD   8
#define HDIM    32
#define NLVL    4
#define NPTS    4
#define NKEY    299520

__device__ __forceinline__ __bf16 f2bf(float f) {
  __hip_bfloat16 h = __float2bfloat16(f);
  return *reinterpret_cast<__bf16*>(&h);
}
__device__ __forceinline__ float bf2f(__bf16 b) {
  __hip_bfloat16 h = *reinterpret_cast<__hip_bfloat16*>(&b);
  return __bfloat162float(h);
}

// squeeze row permutation: value_sq[row] = value[squeeze_src_row(row)]
// per level (s0,s1,s2): dst local = (i0*s2+i2)*s1+i1 ; src local = (i0*s1+i1)*s2+i2
__device__ __forceinline__ int squeeze_src_row(int r) {
  if (r < 262144) {                 // (128,128,16) block = 2048
    int i0 = r >> 11; int rem = r & 2047; int i2 = rem >> 7; int i1 = rem & 127;
    return (((i0 << 7) + i1) << 4) + i2;
  } else if (r < 294912) {          // (64,64,8) block = 512
    int rr = r - 262144;
    int i0 = rr >> 9; int rem = rr & 511; int i2 = rem >> 6; int i1 = rem & 63;
    return 262144 + (((i0 << 6) + i1) << 3) + i2;
  } else if (r < 299008) {          // (32,32,4) block = 128
    int rr = r - 294912;
    int i0 = rr >> 7; int rem = rr & 127; int i2 = rem >> 5; int i1 = rem & 31;
    return 294912 + (((i0 << 5) + i1) << 2) + i2;
  } else {                          // (16,16,2) block = 32
    int rr = r - 299008;
    int i0 = rr >> 5; int rem = rr & 31; int i2 = rem >> 4; int i1 = rem & 15;
    return 299008 + (((i0 << 4) + i1) << 1) + i2;
  }
}

// ---------------------------------------------------------------------------
// WMMA GEMM:  D[M x NTOT] = A[M x 256] * Bt^T + bias (+ identity)
//   A: bf16 row-major ld=256 (or fp32 'value' rows with squeeze permute)
//   Bt: bf16, N x 256 (transposed weight) -> B fragment = contiguous 32B/lane
//   8 waves/block; wave tile 16 x 64; block tile (8/WCOLS*16) x (WCOLS*64)
// ---------------------------------------------------------------------------
template<int NTOT, int WCOLS, bool PERMUTE_A, bool OUT_BF16, bool ADD_IDENTITY>
__global__ __launch_bounds__(256)
void gemm_bf16_kernel(const void* __restrict__ Asrc,
                      const __bf16* __restrict__ Bt,
                      const float* __restrict__ bias,
                      const float* __restrict__ identity,
                      void* __restrict__ outp)
{
  constexpr int WROWS = 8 / WCOLS;
  constexpr int BM    = WROWS * 16;
  constexpr int LDA   = 264;                    // 256 + 8 bf16 pad (anti-bank-conflict)
  __shared__ unsigned short ldsraw[BM * LDA];
  __bf16* ldsA = reinterpret_cast<__bf16*>(ldsraw);

  const int tid   = threadIdx.x;
  const int mBase = blockIdx.x * BM;

  // ---- stage A tile (full K=256) into LDS as bf16 ----
  if (PERMUTE_A) {
    const float* A = (const float*)Asrc;
    for (int c = tid; c < BM * 64; c += 256) {          // 64 float4 per row
      int r = c >> 6, cc = c & 63;
      int src = squeeze_src_row(mBase + r);
      float4 f = reinterpret_cast<const float4*>(A + (size_t)src * 256)[cc];
      __bf16* d = ldsA + r * LDA + cc * 4;
      d[0] = f2bf(f.x); d[1] = f2bf(f.y); d[2] = f2bf(f.z); d[3] = f2bf(f.w);
    }
  } else {
    const __bf16* A = (const __bf16*)Asrc;
    for (int c = tid; c < BM * 32; c += 256) {          // 32 x 16B per row
      int r = c >> 5, cc = c & 31;
      *reinterpret_cast<uint4*>(ldsA + r * LDA + cc * 8) =
          *reinterpret_cast<const uint4*>(A + (size_t)(mBase + r) * 256 + cc * 8);
    }
  }
  __syncthreads();

  const int wave  = tid >> 5;
  const int lane  = tid & 31;
  const int half  = lane >> 4;     // lane group (K-halves of fragments)
  const int l16   = lane & 15;
  const int waveM = wave / WCOLS;
  const int waveN = wave % WCOLS;
  const int mRow  = waveM * 16;
  const int nBase = waveN * 64;

  // ---- C init = bias broadcast along M ----
  v8f acc[4];
#pragma unroll
  for (int t = 0; t < 4; ++t) {
    float bv = bias[nBase + t * 16 + l16];
#pragma unroll
    for (int j = 0; j < 8; ++j) acc[t][j] = bv;
  }

  // ---- K loop: 8 x (K=32) WMMA steps, 4 N-subtiles each ----
#pragma unroll
  for (int k32 = 0; k32 < 256; k32 += 32) {
    // A fragment (16x32, contiguous K order matching B):
    //   elems 0..7  -> K = k32 + half*16 + 0..7
    //   elems 8..15 -> K = k32 + half*16 + 8..15
    const __bf16* ar = ldsA + (mRow + l16) * LDA + k32 + half * 16;
    v8bf lo = *reinterpret_cast<const v8bf*>(ar);
    v8bf hi = *reinterpret_cast<const v8bf*>(ar + 8);
    v16bf af = __builtin_shufflevector(lo, hi, 0,1,2,3,4,5,6,7,8,9,10,11,12,13,14,15);
#pragma unroll
    for (int t = 0; t < 4; ++t) {
      // B fragment (32x16): lane n = l16, elems i -> K = k32 + half*16 + i
      const __bf16* br = Bt + (size_t)(nBase + t * 16 + l16) * 256 + k32 + half * 16;
      v16bf bf = *reinterpret_cast<const v16bf*>(br);
      acc[t] = __builtin_amdgcn_wmma_f32_16x16x32_bf16(
          false, af, false, bf, (short)0, acc[t], false, false);
    }
  }

  // ---- epilogue: D vgpr j -> row = j + half*8 ; col = l16 (+subtile) ----
  const int rowBase = mBase + mRow + half * 8;
#pragma unroll
  for (int t = 0; t < 4; ++t) {
    int col = nBase + t * 16 + l16;
#pragma unroll
    for (int j = 0; j < 8; ++j) {
      int row = rowBase + j;
      float vv = acc[t][j];
      if (ADD_IDENTITY) vv += identity[(size_t)row * 256 + col];
      if (OUT_BF16)
        ((__bf16*)outp)[(size_t)row * NTOT + col] = f2bf(vv);
      else
        ((float*)outp)[(size_t)row * NTOT + col] = vv;
    }
  }
}

// ---------------------------------------------------------------------------
// Prep kernels
// ---------------------------------------------------------------------------
__global__ void transpose_w_bf16(const float* __restrict__ W, __bf16* __restrict__ Wt, int N) {
  int idx = blockIdx.x * 256 + threadIdx.x;   // grid covers 256*N exactly
  int k = idx / N, n = idx % N;
  Wt[(size_t)n * 256 + k] = f2bf(W[(size_t)k * N + n]);
}

__global__ void prep_q_bf16(const float* __restrict__ query, const float* __restrict__ qpos,
                            __bf16* __restrict__ qbf) {
  int i = blockIdx.x * 256 + threadIdx.x;     // grid covers NQ*E exactly
  qbf[i] = f2bf(query[i] + qpos[i]);
}

__global__ void prep_rp2(const float* __restrict__ rp, float* __restrict__ rp2) {
  int i = blockIdx.x * 256 + threadIdx.x;     // grid covers NQ*NLVL exactly
  int l = i & 3;
  const float s2inv[4] = {1.f/16.f, 1.f/8.f, 1.f/4.f, 1.f/2.f};
  float r0 = rp[i * 3 + 0], r1 = rp[i * 3 + 1], r2 = rp[i * 3 + 2];
  rp2[i * 2 + 0] = r1;
  rp2[i * 2 + 1] = r2 + (r0 - 0.5f) * s2inv[l];
}

// ---------------------------------------------------------------------------
// Sampler: one wave per (q, h); lanes = 32 head dims.
//   fused softmax over 16 (level,point) logits + 16 bilinear samples.
//   Each corner fetch is a coalesced 64B line from the bf16 v table.
// ---------------------------------------------------------------------------
__device__ __forceinline__ float fetchv(const __bf16* __restrict__ vtab, int start,
                                        int H, int W, int yi, int xi, int h, int lane) {
  bool valid = (yi >= 0) & (yi < H) & (xi >= 0) & (xi < W);
  int yc = yi < 0 ? 0 : (yi > H - 1 ? H - 1 : yi);
  int xc = xi < 0 ? 0 : (xi > W - 1 ? W - 1 : xi);
  size_t row = (size_t)start + (size_t)yc * W + xc;
  float val = bf2f(vtab[row * 256 + (size_t)h * 32 + lane]);
  return valid ? val : 0.f;
}

__global__ __launch_bounds__(256)
void sampler_kernel(const __bf16* __restrict__ vtab, const float* __restrict__ offv,
                    const float* __restrict__ logits, const float* __restrict__ rp2,
                    __bf16* __restrict__ sampled)
{
  const int tid  = threadIdx.x;
  const int lane = tid & 31;
  const int gw   = blockIdx.x * 8 + (tid >> 5);   // 65536 waves total
  const int q = gw >> 3, h = gw & 7;

  // softmax over the 16 (l,p) logits for this (q,h)
  const float* lg = logits + (size_t)(q * 8 + h) * 16;
  float mx = lg[0];
#pragma unroll
  for (int i = 1; i < 16; ++i) mx = fmaxf(mx, lg[i]);
  float w[16], s = 0.f;
#pragma unroll
  for (int i = 0; i < 16; ++i) { w[i] = __expf(lg[i] - mx); s += w[i]; }
  const float inv = 1.0f / s;

  const int Hs[4] = {2048, 512, 128, 32};
  const int Ws[4] = {128, 64, 32, 16};
  const int St[4] = {0, 262144, 294912, 299008};

  // off columns for (q, h): base h*32, then (l*4+p)*2 + {x,y}
  const float* offq = offv + (size_t)q * 256 + h * 32;

  float acc = 0.f;
#pragma unroll
  for (int l = 0; l < 4; ++l) {
    const float rx = rp2[(q * 4 + l) * 2 + 0];
    const float ry = rp2[(q * 4 + l) * 2 + 1];
    const int H = Hs[l], W = Ws[l], start = St[l];
    // x = (rp2x + off/W)*W - 0.5 = rp2x*W + off - 0.5  (norm cancels)
    const float fx = rx * (float)W - 0.5f;
    const float fy = ry * (float)H - 0.5f;
#pragma unroll
    for (int p = 0; p < 4; ++p) {
      float x = fx + offq[(l * 4 + p) * 2 + 0];
      float y = fy + offq[(l * 4 + p) * 2 + 1];
      float x0f = floorf(x), y0f = floorf(y);
      float wx = x - x0f, wy = y - y0f;
      int x0 = (int)x0f, y0 = (int)y0f;
      float v00 = fetchv(vtab, start, H, W, y0,     x0,     h, lane);
      float v01 = fetchv(vtab, start, H, W, y0,     x0 + 1, h, lane);
      float v10 = fetchv(vtab, start, H, W, y0 + 1, x0,     h, lane);
      float v11 = fetchv(vtab, start, H, W, y0 + 1, x0 + 1, h, lane);
      float bil = (1.f - wy) * ((1.f - wx) * v00 + wx * v01)
                +        wy  * ((1.f - wx) * v10 + wx * v11);
      acc += (w[l * 4 + p] * inv) * bil;
    }
  }
  sampled[(size_t)q * 256 + h * 32 + lane] = f2bf(acc);
}

// ---------------------------------------------------------------------------
// Host launcher
// ---------------------------------------------------------------------------
extern "C" void kernel_launch(void* const* d_in, const int* in_sizes, int n_in,
                              void* d_out, int out_size, void* d_ws, size_t ws_size,
                              hipStream_t stream) {
  const float* query  = (const float*)d_in[0];
  const float* value  = (const float*)d_in[1];
  const float* qpos   = (const float*)d_in[2];
  const float* refpts = (const float*)d_in[3];
  const float* W_off  = (const float*)d_in[4];
  const float* b_off  = (const float*)d_in[5];
  const float* W_attn = (const float*)d_in[6];
  const float* b_attn = (const float*)d_in[7];
  const float* W_val  = (const float*)d_in[8];
  const float* b_val  = (const float*)d_in[9];
  const float* W_out  = (const float*)d_in[10];
  const float* b_out  = (const float*)d_in[11];
  float* out = (float*)d_out;

  // workspace carve-up (256B aligned slabs)
  char* ws = (char*)d_ws;
  size_t o = 0;
  auto take = [&](size_t bytes) -> char* {
    char* p = ws + o;
    o += (bytes + 255) & ~(size_t)255;
    return p;
  };
  __bf16* WvalT  = (__bf16*)take(256 * 256 * 2);
  __bf16* WoffT  = (__bf16*)take(256 * 256 * 2);
  __bf16* WattnT = (__bf16*)take(256 * 128 * 2);
  __bf16* WoutT  = (__bf16*)take(256 * 256 * 2);
  __bf16* qbf    = (__bf16*)take((size_t)NQ * 256 * 2);
  float*  rp2    = (float*) take((size_t)NQ * NLVL * 2 * 4);
  float*  offF   = (float*) take((size_t)NQ * 256 * 4);
  float*  logits = (float*) take((size_t)NQ * 128 * 4);
  __bf16* sampled= (__bf16*)take((size_t)NQ * 256 * 2);
  __bf16* vbf    = (__bf16*)take((size_t)NKEY * 256 * 2);   // ~153 MB, L2-friendly

  // --- prep ---
  transpose_w_bf16<<<256, 256, 0, stream>>>(W_val,  WvalT,  256);
  transpose_w_bf16<<<256, 256, 0, stream>>>(W_off,  WoffT,  256);
  transpose_w_bf16<<<128, 256, 0, stream>>>(W_attn, WattnT, 128);
  transpose_w_bf16<<<256, 256, 0, stream>>>(W_out,  WoutT,  256);
  prep_q_bf16<<<(NQ * 256) / 256, 256, 0, stream>>>(query, qpos, qbf);
  prep_rp2<<<(NQ * NLVL) / 256, 256, 0, stream>>>(refpts, rp2);

  // --- value projection (squeeze fused into A stager), 299520/32 tiles ---
  gemm_bf16_kernel<256, 4, true, true, false>
      <<<NKEY / 32, 256, 0, stream>>>(value, WvalT, b_val, nullptr, vbf);

  // --- q-side GEMMs ---
  gemm_bf16_kernel<256, 4, false, false, false>
      <<<NQ / 32, 256, 0, stream>>>(qbf, WoffT, b_off, nullptr, offF);
  gemm_bf16_kernel<128, 2, false, false, false>
      <<<NQ / 64, 256, 0, stream>>>(qbf, WattnT, b_attn, nullptr, logits);

  // --- deformable sampling (fused softmax) ---
  sampler_kernel<<<NQ, 256, 0, stream>>>(vbf, offF, logits, rp2, sampled);

  // --- output projection + residual ---
  gemm_bf16_kernel<256, 4, false, false, true>
      <<<NQ / 32, 256, 0, stream>>>(sampled, WoutT, b_out, query, out);

  (void)in_sizes; (void)n_in; (void)out_size; (void)ws_size;
}
